// MultiScaleAttention_67267777790164
// MI455X (gfx1250) — compile-verified, matrix operand order checked by reading, not run on hardware
//
#include <hip/hip_runtime.h>

typedef __attribute__((ext_vector_type(16))) __bf16 v16b;
typedef __attribute__((ext_vector_type(8)))  float  v8f;
typedef __attribute__((ext_vector_type(4)))  int    v4i;

#define DEV __device__ __forceinline__
#define AS1 __attribute__((address_space(1)))
#define AS3 __attribute__((address_space(3)))

#if defined(__HIP_DEVICE_COMPILE__) && __has_builtin(__builtin_amdgcn_global_load_async_to_lds_b128)
#define HAVE_ASYNC_LDS 1
#else
#define HAVE_ASYNC_LDS 0
#endif

constexpr int BATCH   = 8;
constexpr int NTOK    = 1569;
constexpr int DIM     = 768;
constexpr int HEADS   = 8;
constexpr int HD      = 96;
constexpr int NQ      = 1569;
constexpr int NK      = 393;
constexpr int M_ROWS  = BATCH * NTOK;   // 12552
constexpr int VT_STRIDE_G = 400;        // global Vt col stride (padded)

DEV unsigned short f2b(float f) {
    unsigned u = __float_as_uint(f);
    u += 0x7FFFu + ((u >> 16) & 1u);    // round-to-nearest-even
    return (unsigned short)(u >> 16);
}
DEV float b2f(unsigned short h) { return __uint_as_float(((unsigned)h) << 16); }

DEV v8f zero8() { v8f z = {0.f,0.f,0.f,0.f,0.f,0.f,0.f,0.f}; return z; }

// 16-byte global -> LDS copy; CDNA5 async path when available.
DEV void async_copy16(const void* g, void* l) {
#if HAVE_ASYNC_LDS
    __builtin_amdgcn_global_load_async_to_lds_b128(
        (AS1 v4i*)(void*)g, (AS3 v4i*)l, 0, 0);
#else
    *(uint4*)l = *(const uint4*)g;
#endif
}
template<int N>
DEV void async_wait_le() {
#if HAVE_ASYNC_LDS
#if __has_builtin(__builtin_amdgcn_s_wait_asynccnt)
    __builtin_amdgcn_s_wait_asynccnt(N);
#else
    asm volatile("s_wait_asynccnt %0" :: "i"(N) : "memory");
#endif
#endif
}

// Load a 16x32 A-style bf16 fragment for one lane:
// elements 0..7  = p[0..7]   (K = kb .. kb+7)
// elements 8..15 = p[16..23] (K = kb+16 .. kb+23)
DEV v16b load_frag(const unsigned short* p) {
    union { uint4 q[2]; v16b v; } u;
    u.q[0] = *(const uint4*)(p);
    u.q[1] = *(const uint4*)(p + 16);
    return u.v;
}

// ---------------------------------------------------------------- converts
__global__ void cvt_f32_bf16(const float* __restrict__ in,
                             unsigned short* __restrict__ out, int n) {
    int i = blockIdx.x * blockDim.x + threadIdx.x;
    if (i < n) out[i] = f2b(in[i]);
}

// in: f32 [K, N] row-major -> out: bf16 [N, K] (transposed, so GEMM B tiles
// stage with contiguous b128 copies just like A tiles)
__global__ void cvt_transpose_bf16(const float* __restrict__ in,
                                   unsigned short* __restrict__ out,
                                   int K, int N) {
    int i = blockIdx.x * blockDim.x + threadIdx.x;
    if (i < K * N) {
        int n = i / K, k = i - n * K;
        out[i] = f2b(in[(size_t)k * N + n]);
    }
}

// ---------------------------------------------------------------- GEMM
// C[M,N](f32 or bf16) = A[M,K](bf16) @ Bt[N,K](bf16)^T + bias
// block = 256 threads (8 waves), block tile 256x128, wave tile 64x64,
// double-buffered async-to-LDS staging (6 async b128 per wave per k-step).
template<bool OUT_BF16>
__global__ void gemm_wmma(const unsigned short* __restrict__ A,
                          const unsigned short* __restrict__ Bt,
                          const float* __restrict__ bias,
                          void* __restrict__ Cout,
                          int M, int N, int K) {
    __shared__ alignas(16) unsigned short sA[2][256 * 40];
    __shared__ alignas(16) unsigned short sB[2][128 * 40];

    const int tid  = threadIdx.x;
    const int lane = tid & 31;
    const int wid  = tid >> 5;
    const int wm   = wid & 3;       // 4 waves along M (64 rows each)
    const int wn   = wid >> 2;      // 2 waves along N (64 cols each)
    const int m0   = blockIdx.y * 256;
    const int n0   = blockIdx.x * 128;

    const int frow = lane & 15;
    const int kb   = (lane >> 4) * 8;

    auto stageA = [&](int buf, int kc) {
        #pragma unroll
        for (int c = tid; c < 512; c += 256) {
            int row = c >> 1, half = (c & 1) * 16;
            int grow = m0 + row; if (grow > M - 1) grow = M - 1;
            const unsigned short* src = A + (size_t)grow * K + kc + half;
            unsigned short* dst = &sA[buf][row * 40 + half];
            async_copy16(src, dst);
            async_copy16(src + 8, dst + 8);
        }
    };
    auto stageB = [&](int buf, int kc) {
        int row = tid >> 1, half = (tid & 1) * 16;
        int grow = n0 + row; if (grow > N - 1) grow = N - 1;
        const unsigned short* src = Bt + (size_t)grow * K + kc + half;
        unsigned short* dst = &sB[buf][row * 40 + half];
        async_copy16(src, dst);
        async_copy16(src + 8, dst + 8);
    };

    v8f acc[4][4];
    #pragma unroll
    for (int i = 0; i < 4; i++)
        #pragma unroll
        for (int j = 0; j < 4; j++) acc[i][j] = zero8();

    const int T = K / 32;
    stageA(0, 0);
    stageB(0, 0);
    for (int t = 0; t < T; t++) {
        const int cur = t & 1;
        if (t + 1 < T) {
            stageA(cur ^ 1, (t + 1) * 32);
            stageB(cur ^ 1, (t + 1) * 32);
            async_wait_le<6>();      // in-order completion => tile t staged
        } else {
            async_wait_le<0>();
        }
        __syncthreads();

        v16b aF[4];
        #pragma unroll
        for (int s = 0; s < 4; s++)
            aF[s] = load_frag(&sA[cur][(wm * 64 + s * 16 + frow) * 40 + kb]);
        #pragma unroll
        for (int sn = 0; sn < 4; sn++) {
            v16b bF = load_frag(&sB[cur][(wn * 64 + sn * 16 + frow) * 40 + kb]);
            #pragma unroll
            for (int sm = 0; sm < 4; sm++)
                acc[sm][sn] = __builtin_amdgcn_wmma_f32_16x16x32_bf16(
                    false, aF[sm], false, bF, (short)0, acc[sm][sn], false, false);
        }
        __syncthreads();   // done reading buf `cur` before it is restaged
    }

    const int rbase = (lane >> 4) * 8;
    #pragma unroll
    for (int sm = 0; sm < 4; sm++) {
        #pragma unroll
        for (int sn = 0; sn < 4; sn++) {
            int col = n0 + wn * 64 + sn * 16 + frow;
            float bv = bias[col];
            #pragma unroll
            for (int r = 0; r < 8; r++) {
                int row = m0 + wm * 64 + sm * 16 + rbase + r;
                if (row < M) {
                    float v = acc[sm][sn][r] + bv;
                    if (OUT_BF16)
                        ((unsigned short*)Cout)[(size_t)row * N + col] = f2b(v);
                    else
                        ((float*)Cout)[(size_t)row * N + col] = v;
                }
            }
        }
    }
}

// ---------------------------------------------------------------- depthwise conv pool + layernorm
// One wave per output token. qkv: [M_ROWS, 3*DIM] bf16. Output bf16:
//   transposed==0: out[(bh*Ntok + n)*96 + d]        (row-major, for Q/K)
//   transposed==1: out[(bh*96 + d)*400 + n]         (Vt, padded col stride)
__global__ void pool_ln(const unsigned short* __restrict__ qkv,
                        const float* __restrict__ wconv,
                        const float* __restrict__ gamma,
                        const float* __restrict__ beta,
                        unsigned short* __restrict__ out,
                        int slice, int OT, int OH, int OW,
                        int sT, int sH, int sW, int transposed) {
    const int Ntok_out = 1 + OT * OH * OW;
    const int total = BATCH * HEADS * Ntok_out;
    const int wgid = blockIdx.x * (blockDim.x >> 5) + (threadIdx.x >> 5);
    if (wgid >= total) return;

    const int lane = threadIdx.x & 31;
    int b   = wgid / (HEADS * Ntok_out);
    int rem = wgid - b * HEADS * Ntok_out;
    int h   = rem / Ntok_out;
    int n   = rem - h * Ntok_out;

    float v[3];
    #pragma unroll
    for (int c = 0; c < 3; c++) {
        int d = lane + c * 32;
        int colbase = slice * DIM + h * HD + d;
        float acc = 0.f;
        if (n == 0) {
            acc = b2f(qkv[(size_t)(b * NTOK) * (3 * DIM) + colbase]);
        } else {
            int p  = n - 1;
            int t  = p / (OH * OW);
            int r2 = p - t * (OH * OW);
            int y  = r2 / OW, x = r2 - y * OW;
            for (int dt = 0; dt < 3; dt++) {
                int it = t * sT + dt - 1;
                if (it < 0 || it >= 8) continue;
                for (int dy = 0; dy < 3; dy++) {
                    int iy = y * sH + dy - 1;
                    if (iy < 0 || iy >= 14) continue;
                    for (int dx = 0; dx < 3; dx++) {
                        int ix = x * sW + dx - 1;
                        if (ix < 0 || ix >= 14) continue;
                        int irow = b * NTOK + 1 + (it * 14 + iy) * 14 + ix;
                        acc += b2f(qkv[(size_t)irow * (3 * DIM) + colbase]) *
                               wconv[((dt * 3 + dy) * 3 + dx) * HD + d];
                    }
                }
            }
        }
        v[c] = acc;
    }
    // layernorm over 96 channels (wave32 reduction)
    float s  = v[0] + v[1] + v[2];
    float sq = v[0]*v[0] + v[1]*v[1] + v[2]*v[2];
    #pragma unroll
    for (int m = 16; m >= 1; m >>= 1) {
        s  += __shfl_xor(s,  m, 32);
        sq += __shfl_xor(sq, m, 32);
    }
    float mu  = s * (1.f / 96.f);
    float var = sq * (1.f / 96.f) - mu * mu;
    float inv = rsqrtf(var + 1e-5f);
    #pragma unroll
    for (int c = 0; c < 3; c++) {
        int d = lane + c * 32;
        float o = (v[c] - mu) * inv * gamma[d] + beta[d];
        if (transposed)
            out[((size_t)(b * HEADS + h) * HD + d) * VT_STRIDE_G + n] = f2b(o);
        else
            out[((size_t)(b * HEADS + h) * Ntok_out + n) * HD + d] = f2b(o);
    }
}

// ---------------------------------------------------------------- fused attention
// Per workgroup: one (b,h) and a 128-query tile; 8 waves x 16 query rows.
// LDS: [K rows | reused for Vt] + S(f32, in-place compacted to bf16 P) + row sums.
constexpr int KSTRIDE = 104;                         // ushorts per K row
constexpr int VSTRIDE = 424;                         // ushorts per Vt row
constexpr int SSTRIDE = 404;                         // dwords per S row
constexpr int KV_BYTES = 400 * KSTRIDE * 2;          // 83200
constexpr int S_OFF    = KV_BYTES;
constexpr int S_BYTES  = 8 * 16 * SSTRIDE * 4;       // 206848
constexpr int SUM_OFF  = S_OFF + S_BYTES;            // 290048
constexpr int SMEM_BYTES = SUM_OFF + 8 * 16 * 4;     // 290560  (<= 320KB LDS/WGP)

__global__ void attn_fused(const unsigned short* __restrict__ Qg,
                           const unsigned short* __restrict__ Kg,
                           const unsigned short* __restrict__ Vtg,
                           unsigned short* __restrict__ Og) {
    extern __shared__ char smem[];
    unsigned short* sKV  = (unsigned short*)smem;
    float*          Sf   = (float*)(smem + S_OFF);
    float*          sums = (float*)(smem + SUM_OFF);

    const int tid  = threadIdx.x;
    const int lane = tid & 31;
    const int wid  = tid >> 5;
    const int bh   = blockIdx.y;
    const int b    = bh >> 3, h = bh & 7;
    const int qbase = blockIdx.x * 128 + wid * 16;
    const float scale = 0.10206207261596575f;        // 96^-0.5

    const int frow = lane & 15;
    const int kb   = (lane >> 4) * 8;

    // ---- phase 1: K -> LDS (rows >= NK zeroed), async b128 chunks
    for (int c = tid; c < 400 * 12; c += 256) {
        int n = c / 12;
        int d = (c - n * 12) * 8;
        if (n < NK) {
            async_copy16(Kg + ((size_t)bh * NK + n) * HD + d,
                         sKV + n * KSTRIDE + d);
        } else {
            uint4 z; z.x = z.y = z.z = z.w = 0u;
            *(uint4*)(sKV + n * KSTRIDE + d) = z;
        }
    }
    async_wait_le<0>();
    __syncthreads();

    // ---- phase 2: S = scale * Q K^T  (per-wave 16 rows x 400 cols)
    v16b qF[3];
    {
        int row = qbase + frow; if (row > NQ - 1) row = NQ - 1;
        const unsigned short* qp = Qg + ((size_t)bh * NQ + row) * HD;
        #pragma unroll
        for (int kk = 0; kk < 3; kk++) qF[kk] = load_frag(qp + kk * 32 + kb);
    }
    float* Sw = Sf + wid * 16 * SSTRIDE;
    for (int j = 0; j < 25; j++) {
        v8f acc = zero8();
        const unsigned short* kp = sKV + (j * 16 + frow) * KSTRIDE + kb;
        #pragma unroll
        for (int kk = 0; kk < 3; kk++) {
            v16b bF = load_frag(kp + kk * 32);
            acc = __builtin_amdgcn_wmma_f32_16x16x32_bf16(
                false, qF[kk], false, bF, (short)0, acc, false, false);
        }
        int col = j * 16 + frow;
        #pragma unroll
        for (int r = 0; r < 8; r++) {
            int rw = (lane >> 4) * 8 + r;
            Sw[rw * SSTRIDE + col] =
                (col < NK) ? acc[r] * scale : -__builtin_inff();
        }
    }
    __syncthreads();   // all waves done reading K

    // ---- phase 3: Vt -> LDS (reuse K region); cols >= NK zeroed
    // full-width async b128 chunks cover cols 0..391
    for (int c = tid; c < 96 * 49; c += 256) {
        int d = c / 49;
        int n = (c - d * 49) * 8;
        async_copy16(Vtg + ((size_t)bh * HD + d) * VT_STRIDE_G + n,
                     sKV + d * VSTRIDE + n);
    }
    // scalar tail: cols 392..415 (one real col + zero padding)
    for (int i = tid; i < 96 * 24; i += 256) {
        int d = i / 24;
        int n = 392 + (i - d * 24);
        sKV[d * VSTRIDE + n] =
            (n < NK) ? Vtg[((size_t)bh * HD + d) * VT_STRIDE_G + n]
                     : (unsigned short)0;
    }

    // ---- phase 4: softmax on own rows; compact P to bf16 in place
    if (lane < 16) {
        float* row = Sw + lane * SSTRIDE;
        float mx = -__builtin_inff();
        for (int i = 0; i < NK; i++) mx = fmaxf(mx, row[i]);
        float sum = 0.f;
        unsigned* prow = (unsigned*)row;
        for (int i = 0; i < 200; i++) {
            float e0 = __expf(row[2 * i]     - mx);
            float e1 = __expf(row[2 * i + 1] - mx);
            sum += e0 + e1;
            prow[i] = ((unsigned)f2b(e1) << 16) | (unsigned)f2b(e0);
        }
        for (int i = 200; i < 208; i++) prow[i] = 0u;   // keys 400..415 = 0
        sums[wid * 16 + lane] = sum;
    }
    async_wait_le<0>();
    __syncthreads();   // V staged + P ready

    // ---- phase 5: O = (P V) / rowsum
    const unsigned short* Pus = (const unsigned short*)Sw;
    v8f oacc[6];
    #pragma unroll
    for (int t = 0; t < 6; t++) oacc[t] = zero8();
    for (int kc = 0; kc < 13; kc++) {
        v16b pF = load_frag(Pus + frow * (SSTRIDE * 2) + kc * 32 + kb);
        #pragma unroll
        for (int t = 0; t < 6; t++) {
            v16b vF = load_frag(sKV + (t * 16 + frow) * VSTRIDE + kc * 32 + kb);
            oacc[t] = __builtin_amdgcn_wmma_f32_16x16x32_bf16(
                false, pF, false, vF, (short)0, oacc[t], false, false);
        }
    }
    #pragma unroll
    for (int r = 0; r < 8; r++) {
        int rw = (lane >> 4) * 8 + r;
        int grow = qbase + rw;
        if (grow < NQ) {
            float inv = 1.f / sums[wid * 16 + rw];
            #pragma unroll
            for (int t = 0; t < 6; t++) {
                int d = t * 16 + frow;
                Og[((size_t)b * NQ + grow) * DIM + h * HD + d] = f2b(oacc[t][r] * inv);
            }
        }
    }
}

// ---------------------------------------------------------------- host
// workspace layout (bytes)
constexpr size_t OFF_XB  = 0;
constexpr size_t OFF_QW  = OFF_XB  + (size_t)M_ROWS * DIM * 2;
constexpr size_t OFF_PW  = OFF_QW  + (size_t)DIM * 3 * DIM * 2;
constexpr size_t OFF_QKV = OFF_PW  + (size_t)DIM * DIM * 2;
constexpr size_t OFF_Q   = OFF_QKV + (size_t)M_ROWS * 3 * DIM * 2;
constexpr size_t OFF_K   = OFF_Q   + (size_t)BATCH * HEADS * NQ * HD * 2;
constexpr size_t OFF_VT  = OFF_K   + (size_t)BATCH * HEADS * NK * HD * 2;
constexpr size_t OFF_O   = OFF_VT  + (size_t)BATCH * HEADS * HD * VT_STRIDE_G * 2;

extern "C" void kernel_launch(void* const* d_in, const int* in_sizes, int n_in,
                              void* d_out, int out_size, void* d_ws, size_t ws_size,
                              hipStream_t stream) {
    (void)in_sizes; (void)n_in; (void)out_size; (void)ws_size;
    const float* x      = (const float*)d_in[0];
    const float* qkv_w  = (const float*)d_in[1];
    const float* qkv_b  = (const float*)d_in[2];
    const float* proj_w = (const float*)d_in[3];
    const float* proj_b = (const float*)d_in[4];
    const float* wq = (const float*)d_in[5];
    const float* gq = (const float*)d_in[6];
    const float* bq = (const float*)d_in[7];
    const float* wk = (const float*)d_in[8];
    const float* gk = (const float*)d_in[9];
    const float* bk = (const float*)d_in[10];
    const float* wv = (const float*)d_in[11];
    const float* gv = (const float*)d_in[12];
    const float* bv = (const float*)d_in[13];

    char* ws = (char*)d_ws;
    unsigned short* xb   = (unsigned short*)(ws + OFF_XB);
    unsigned short* qwb  = (unsigned short*)(ws + OFF_QW);   // bf16 [2304, 768] (qkv_w^T)
    unsigned short* pwb  = (unsigned short*)(ws + OFF_PW);   // bf16 [768, 768]  (proj_w^T)
    unsigned short* qkvb = (unsigned short*)(ws + OFF_QKV);
    unsigned short* qp   = (unsigned short*)(ws + OFF_Q);
    unsigned short* kp   = (unsigned short*)(ws + OFF_K);
    unsigned short* vtp  = (unsigned short*)(ws + OFF_VT);
    unsigned short* ob   = (unsigned short*)(ws + OFF_O);

    // 1. f32 -> bf16 conversions (weights transposed to [N,K])
    int nx  = M_ROWS * DIM;
    int nqw = DIM * 3 * DIM;
    int npw = DIM * DIM;
    cvt_f32_bf16<<<(nx + 255) / 256, 256, 0, stream>>>(x, xb, nx);
    cvt_transpose_bf16<<<(nqw + 255) / 256, 256, 0, stream>>>(qkv_w,  qwb, DIM, 3 * DIM);
    cvt_transpose_bf16<<<(npw + 255) / 256, 256, 0, stream>>>(proj_w, pwb, DIM, DIM);

    // 2. QKV GEMM (bf16 out)
    gemm_wmma<true><<<dim3(3 * DIM / 128, (M_ROWS + 255) / 256), 256, 0, stream>>>(
        xb, qwb, qkv_b, qkvb, M_ROWS, 3 * DIM, DIM);

    // 3. attention pools (depthwise conv + LN)
    {
        int tq = BATCH * HEADS * NQ;
        pool_ln<<<(tq + 7) / 8, 256, 0, stream>>>(qkvb, wq, gq, bq, qp,
                                                  0, 8, 14, 14, 1, 1, 1, 0);
        int tk = BATCH * HEADS * NK;
        pool_ln<<<(tk + 7) / 8, 256, 0, stream>>>(qkvb, wk, gk, bk, kp,
                                                  1, 8, 7, 7, 1, 2, 2, 0);
        pool_ln<<<(tk + 7) / 8, 256, 0, stream>>>(qkvb, wv, gv, bv, vtp,
                                                  2, 8, 7, 7, 1, 2, 2, 1);
    }

    // 4. fused attention (K,V resident in 320KB LDS)
    attn_fused<<<dim3((NQ + 127) / 128, BATCH * HEADS), 256, SMEM_BYTES, stream>>>(
        qp, kp, vtp, ob);

    // 5. output projection (f32 out)
    gemm_wmma<false><<<dim3(DIM / 128, (M_ROWS + 255) / 256), 256, 0, stream>>>(
        ob, pwb, proj_b, (float*)d_out, M_ROWS, DIM, DIM);
}
